// LeviCivitaKANLayerOptimized_54820962566348
// MI455X (gfx1250) — matrix operand order repaired and checked by reading, not compiled, not run on hardware
//
#include <hip/hip_runtime.h>

typedef __attribute__((ext_vector_type(16))) __bf16 v16bf;
typedef __attribute__((ext_vector_type(8)))  __bf16 v8bf;
typedef __attribute__((ext_vector_type(8)))  float  v8f;

constexpr int BATCH = 32768;
constexpr int IN_F  = 1024;
constexpr int OUT_F = 1024;
constexpr int NP    = 4096;

// ---------------------------------------------------------------------------
// Kernel 1: fp32 -> bf16 weight conversion (lw: [OUT,NP], w1: [OUT,IN])
// ---------------------------------------------------------------------------
__global__ __launch_bounds__(256) void k_convert_weights(
    const float* __restrict__ lw, const float* __restrict__ w1,
    __bf16* __restrict__ lwb, __bf16* __restrict__ w1b) {
  int stride = gridDim.x * 256;
  int i0 = blockIdx.x * 256 + threadIdx.x;
  for (int t = i0; t < OUT_F * NP; t += stride) lwb[t] = (__bf16)lw[t];
  for (int t = i0; t < OUT_F * IN_F; t += stride) w1b[t] = (__bf16)w1[t];
}

// ---------------------------------------------------------------------------
// Kernel 2: per batch row: xs = (x - t)/s in LDS, pair gather-multiply -> bf16,
//           and bf16 copy of raw x (base GEMM uses un-normalized x).
// ---------------------------------------------------------------------------
__global__ __launch_bounds__(256) void k_prep(
    const float* __restrict__ x,
    const int* __restrict__ idx_i, const int* __restrict__ idx_j,
    const float* __restrict__ scale, const float* __restrict__ translation,
    __bf16* __restrict__ ppb, __bf16* __restrict__ xb) {
  __shared__ float xs[IN_F];
  const int b = blockIdx.x;
  const float* xrow = x + (size_t)b * IN_F;
  __bf16* xbrow = xb + (size_t)b * IN_F;
  for (int i = threadIdx.x; i < IN_F; i += 256) {
    float xv = xrow[i];
    xs[i] = (xv - translation[i]) / scale[i];
    xbrow[i] = (__bf16)xv;
  }
  __syncthreads();
  __bf16* pprow = ppb + (size_t)b * NP;
  for (int p = threadIdx.x; p < NP; p += 256) {
    pprow[p] = (__bf16)(xs[idx_i[p]] * xs[idx_j[p]]);
  }
}

// ---------------------------------------------------------------------------
// Kernel 3: C[b,o] = sum_k pp[b,k]*lw[o,k] + sum_k x[b,k]*w1[o,k]  (fp32 acc)
// Workgroup: 256 threads = 8 waves. Tile: M=128, N=256.
// Wave (mw in 0..1, nw in 0..3) computes a 64x64 subtile = 4x4 WMMA tiles
// -> 16 v_wmma per 32-wide K step against 8 B-fragment b128 loads
// (0.019 B/FLOP of L2 fragment traffic). Double-buffered LDS A tile,
// one barrier per step; next tile fetched to registers during compute.
// B (weights, ~10 MB bf16 total) served from L2 directly.
// ---------------------------------------------------------------------------
#define LDA_PAD 40  // 32 bf16 + 8 pad -> 80B row stride, conflict-free frags

__global__ __launch_bounds__(256) void k_gemm_wmma(
    const __bf16* __restrict__ App,  // [BATCH, NP]
    const __bf16* __restrict__ Wlw,  // [OUT_F, NP]
    const __bf16* __restrict__ Axb,  // [BATCH, IN_F]
    const __bf16* __restrict__ Ww1,  // [OUT_F, IN_F]
    float* __restrict__ out) {       // [BATCH, OUT_F]
  __shared__ __bf16 ldsA[2][128 * LDA_PAD];

  const int tid  = threadIdx.x;
  const int wave = tid >> 5;
  const int lane = tid & 31;
  const int mw   = wave & 1;   // 0..1 -> M offset 0/64
  const int nw   = wave >> 1;  // 0..3 -> N offset 0/64/128/192

  // consecutive blocks share the A tile (4 N-blocks) for L2 reuse
  const int nblk = blockIdx.x & 3;          // OUT_F/256 = 4
  const int mblk = blockIdx.x >> 2;         // BATCH/128 = 256
  const int row0 = mblk * 128;
  const int col0 = nblk * 256 + nw * 64;

  const int mn   = lane & 15;  // matrix row (A) / col (B,C) index
  const int half = lane >> 4;  // lane half selects K sub-range

  // cooperative A-tile staging: 128 rows x 32 K bf16 = 8KB, 32B per thread
  const int srow   = tid >> 2;        // 0..63 (and +64 for second chunk)
  const int schunk = (tid & 3) * 8;   // 0,8,16,24 (bf16 elems)

  v8f acc[4][4] = {};
  int p = 0;  // LDS buffer selector

  auto kphase = [&](const __bf16* __restrict__ Ag,
                    const __bf16* __restrict__ Wg, int K) {
    const __bf16* asrc0 = Ag + (size_t)(row0 + srow) * K + schunk;
    const __bf16* asrc1 = Ag + (size_t)(row0 + srow + 64) * K + schunk;
    // prologue: stage K-tile 0
    *(v8bf*)&ldsA[p][srow * LDA_PAD + schunk] = *(const v8bf*)asrc0;
    *(v8bf*)&ldsA[p][(srow + 64) * LDA_PAD + schunk] = *(const v8bf*)asrc1;
    __syncthreads();

    for (int kb = 0; kb < K; kb += 32) {
      const bool more = (kb + 32) < K;
      v8bf nxt0, nxt1;
      if (more) {                                  // prefetch next tile
        nxt0 = *(const v8bf*)(asrc0 + kb + 32);
        nxt1 = *(const v8bf*)(asrc1 + kb + 32);
      }

      // B fragments: lane n = mn, K = kb + half*16 .. +15 (contiguous 32B)
      v16bf bfrag[4];
#pragma unroll
      for (int nt = 0; nt < 4; ++nt) {
        bfrag[nt] =
            *(const v16bf*)(Wg + (size_t)(col0 + nt * 16 + mn) * K + kb +
                            half * 16);
      }
#pragma unroll
      for (int mt = 0; mt < 4; ++mt) {
        // A fragment: lane m = mn; K = half*16 .. +15 (contiguous)
        const __bf16* arow = &ldsA[p][(mw * 64 + mt * 16 + mn) * LDA_PAD];
        v8bf lo = *(const v8bf*)(arow + half * 16);
        v8bf hi = *(const v8bf*)(arow + half * 16 + 8);
        v16bf afrag;
#pragma unroll
        for (int e = 0; e < 8; ++e) { afrag[e] = lo[e]; afrag[e + 8] = hi[e]; }
#pragma unroll
        for (int nt = 0; nt < 4; ++nt) {
          acc[mt][nt] = __builtin_amdgcn_wmma_f32_16x16x32_bf16(
              false, afrag, false, bfrag[nt], (short)0, acc[mt][nt],
              false, false);
        }
      }
      // stage next tile into the other buffer; wmma already consumed this
      // buffer's ds_loads, so one barrier per step is sufficient.
      if (more) {
        *(v8bf*)&ldsA[p ^ 1][srow * LDA_PAD + schunk] = nxt0;
        *(v8bf*)&ldsA[p ^ 1][(srow + 64) * LDA_PAD + schunk] = nxt1;
      }
      __syncthreads();
      p ^= 1;
    }
  };

  kphase(App, Wlw, NP);    // pair-product GEMM, K = 4096
  kphase(Axb, Ww1, IN_F);  // base GEMM, K = 1024

  // C/D layout: lane -> (n = lane%16), reg r -> (m = r + 8*(lane/16))
#pragma unroll
  for (int mt = 0; mt < 4; ++mt) {
    const int rbase = row0 + mw * 64 + mt * 16 + half * 8;
#pragma unroll
    for (int nt = 0; nt < 4; ++nt) {
      const int col = col0 + nt * 16 + mn;
#pragma unroll
      for (int r = 0; r < 8; ++r)
        out[(size_t)(rbase + r) * OUT_F + col] = acc[mt][nt][r];
    }
  }
}

// ---------------------------------------------------------------------------
// Kernel 4: LayerNorm in-place on out. One wave32 per row (1024 elems).
// ---------------------------------------------------------------------------
__global__ __launch_bounds__(256) void k_layernorm(
    float* __restrict__ out,
    const float* __restrict__ gamma, const float* __restrict__ beta) {
  const int wave = threadIdx.x >> 5;
  const int lane = threadIdx.x & 31;
  const int row  = blockIdx.x * 8 + wave;
  float* p = out + (size_t)row * OUT_F;

  float v[32];
  float s = 0.f;
#pragma unroll
  for (int i = 0; i < 32; ++i) { v[i] = p[lane + i * 32]; s += v[i]; }
#pragma unroll
  for (int off = 16; off; off >>= 1) s += __shfl_xor(s, off, 32);
  const float mu = s * (1.0f / OUT_F);

  float q = 0.f;
#pragma unroll
  for (int i = 0; i < 32; ++i) { float d = v[i] - mu; q += d * d; }
#pragma unroll
  for (int off = 16; off; off >>= 1) q += __shfl_xor(q, off, 32);
  const float rs = rsqrtf(q * (1.0f / OUT_F) + 1e-5f);

#pragma unroll
  for (int i = 0; i < 32; ++i) {
    const int c = lane + i * 32;
    p[c] = (v[i] - mu) * rs * gamma[c] + beta[c];
  }
}

// ---------------------------------------------------------------------------
extern "C" void kernel_launch(void* const* d_in, const int* in_sizes, int n_in,
                              void* d_out, int out_size, void* d_ws,
                              size_t ws_size, hipStream_t stream) {
  (void)in_sizes; (void)n_in; (void)out_size; (void)ws_size;
  const float* x           = (const float*)d_in[0];
  const int*   idx_i       = (const int*)d_in[1];
  const int*   idx_j       = (const int*)d_in[2];
  const float* scale       = (const float*)d_in[3];
  const float* translation = (const float*)d_in[4];
  const float* lw          = (const float*)d_in[5];
  const float* w1          = (const float*)d_in[6];
  const float* gamma       = (const float*)d_in[7];
  const float* beta        = (const float*)d_in[8];
  float* out = (float*)d_out;

  char* ws = (char*)d_ws;
  size_t off = 0;
  auto carve = [&](size_t bytes) -> char* {
    char* p = ws + off;
    off += (bytes + 255) & ~(size_t)255;
    return p;
  };
  __bf16* ppb = (__bf16*)carve((size_t)BATCH * NP * 2);   // 268 MB
  __bf16* xb  = (__bf16*)carve((size_t)BATCH * IN_F * 2); //  67 MB
  __bf16* lwb = (__bf16*)carve((size_t)OUT_F * NP * 2);   // 8.4 MB
  __bf16* w1b = (__bf16*)carve((size_t)OUT_F * IN_F * 2); // 2.1 MB

  k_convert_weights<<<1024, 256, 0, stream>>>(lw, w1, lwb, w1b);
  k_prep<<<BATCH, 256, 0, stream>>>(x, idx_i, idx_j, scale, translation,
                                    ppb, xb);
  k_gemm_wmma<<<(BATCH / 128) * (OUT_F / 256), 256, 0, stream>>>(
      ppb, lwb, xb, w1b, out);
  k_layernorm<<<BATCH / 8, 256, 0, stream>>>(out, gamma, beta);
}
